// MultiQueryAttention_75746043232282
// MI455X (gfx1250) — matrix operand order repaired and checked by reading, not compiled, not run on hardware
//
#include <hip/hip_runtime.h>

// ---------------------------------------------------------------------------
// MQA forward for MI455X (gfx1250, wave32, WMMA + TDM).
//   x  [4,2048,1024] f32, Wq [1024,1024], Wk [64,1024], Wv [64,1024]
//   out[4,2048,1024] f32
// bf16 WMMA everywhere (f32 accumulate). Projection K-loop is unrolled x2 with
// two alternating register sets (software pipeline, no register-copy rotation)
// so VMEM overlaps WMMA with partial s_wait_loadcnt. Flash attention streams
// 32-key K/V tiles into LDS via the Tensor Data Mover (double buffered on
// TENSORcnt) and runs online softmax in f32.
// ---------------------------------------------------------------------------

typedef __attribute__((ext_vector_type(16))) __bf16 v16bf;
typedef __attribute__((ext_vector_type(8)))  __bf16 v8bf;
typedef __attribute__((ext_vector_type(8)))  float  v8f;
typedef __attribute__((ext_vector_type(4)))  unsigned int u32x4;
typedef __attribute__((ext_vector_type(8)))  int i32x8;
typedef __attribute__((ext_vector_type(4)))  int i32x4;

#define DIM_IN 1024
#define NHEAD  16
#define HDIM   64
#define SEQ    2048
#define BATCH  4
#define MROWS  (BATCH * SEQ)   // 8192
#define PROW   40              // P row stride in halves (80B, 16B aligned)
#define KROW   72              // K tile row stride in halves (64 + 8 pad)
#define VROW   40              // V tile row stride in halves (32 + 8 pad)

#if defined(__has_builtin)
#if __has_builtin(__builtin_amdgcn_tensor_load_to_lds) && \
    __has_builtin(__builtin_amdgcn_s_wait_tensorcnt)
#define HAS_TDM 1
#endif
#endif
#ifndef HAS_TDM
#define HAS_TDM 0
#endif

enum { MODE_Q = 0, MODE_KN = 1, MODE_VT = 2 };

static __device__ inline __bf16 f2bf(float f) {
  unsigned u = __builtin_bit_cast(unsigned, f);
  u += 0x7FFFu + ((u >> 16) & 1u);               // round-to-nearest-even
  unsigned short s = (unsigned short)(u >> 16);
  return __builtin_bit_cast(__bf16, s);
}

// A fragment (16x32 bf16, ISA 7.12.2): lane row = l&15; halves 0..7 at
// K = k0 + (l>=16?8:0), halves 8..15 at +16.
static __device__ inline v16bf load_a32(const __bf16* p) {
  v8bf lo = *(const v8bf*)(p);
  v8bf hi = *(const v8bf*)(p + 16);
  v16bf a;
#pragma unroll
  for (int i = 0; i < 8; ++i) { a[i] = lo[i]; a[8 + i] = hi[i]; }
  return a;
}

// B fragment (32x16 bf16): lane col N = l&15; halves 0..7 at K = k0 + (l>=16?8:0), halves 8..15 at +16.
static __device__ inline v16bf load_b32(const __bf16* p) {
  v8bf lo = *(const v8bf*)(p);
  v8bf hi = *(const v8bf*)(p + 16);
  v16bf b;
#pragma unroll
  for (int i = 0; i < 8; ++i) { b[i] = lo[i]; b[8 + i] = hi[i]; }
  return b;
}

// ---- Tensor Data Mover: 2D tile (bf16) global -> LDS, with LDS padding ----
// D# per CDNA5 ISA ch.8: group0 = {count/flags, lds_addr, global_addr, type=2},
// group1 = {data_size/pad | dims | tile dims | dim0 stride}.
static __device__ inline void tdm_load_2d(unsigned lds_off, const void* gaddr,
                                          unsigned td0, unsigned td1,
                                          unsigned long long stride0,
                                          unsigned tile0, unsigned tile1,
                                          unsigned pad_int_code, unsigned pad_amt_code,
                                          int pad_en) {
#if HAS_TDM
  unsigned long long ga = (unsigned long long)gaddr;
  u32x4 g0;
  g0[0] = 1u;                                            // count=1 (valid), user mode
  g0[1] = lds_off;                                       // lds_addr
  g0[2] = (unsigned)(ga & 0xFFFFFFFFu);                  // global_addr[31:0]
  g0[3] = (unsigned)((ga >> 32) & 0x01FFFFFFu) | (2u << 30);  // addr[56:32], type=2
  unsigned w0 = (1u << 16)                               // data_size = 2 bytes
              | (pad_en ? (1u << 20) : 0u)               // pad_enable
              | (pad_int_code << 22) | (pad_amt_code << 25);
  i32x8 g1;
  g1[0] = (int)w0;
  g1[1] = (int)((td0 & 0xFFFFu) << 16);                  // tensor_dim0[15:0]
  g1[2] = (int)((td0 >> 16) | ((td1 & 0xFFFFu) << 16));  // tdim0 hi | tdim1 lo
  g1[3] = (int)((td1 >> 16) | (tile0 << 16));            // tdim1 hi | tile_dim0
  g1[4] = (int)(tile1 & 0xFFFFu);                        // tile_dim1 | tile_dim2=0
  g1[5] = (int)(unsigned)(stride0 & 0xFFFFFFFFu);        // dim0 stride lo
  g1[6] = (int)(unsigned)((stride0 >> 32) & 0xFFFFu);    // dim0 stride hi
  g1[7] = 0;
  i32x4 z = {0, 0, 0, 0};
#if defined(__clang_major__) && (__clang_major__ >= 23)
  i32x8 z8 = {0, 0, 0, 0, 0, 0, 0, 0};
  __builtin_amdgcn_tensor_load_to_lds(g0, g1, z, z, z8, 0);
#else
  __builtin_amdgcn_tensor_load_to_lds(g0, g1, z, z, 0);
#endif
#else
  (void)lds_off; (void)gaddr; (void)td0; (void)td1; (void)stride0;
  (void)tile0; (void)tile1; (void)pad_int_code; (void)pad_amt_code; (void)pad_en;
#endif
}

static __device__ inline void tdm_wait0() {
#if HAS_TDM
  __builtin_amdgcn_s_wait_tensorcnt(0);
#endif
}

#if !HAS_TDM
// Cooperative fallback: one wave copies a padded tile via b128 loads/stores.
static __device__ inline void stage_tile(__bf16* dst, const __bf16* g, int rows,
                                         int rowhalves, int gstride, int lstride,
                                         int lane) {
  int cpr = rowhalves >> 3;            // 16B chunks per row
  int total = rows * cpr;
  for (int idx = lane; idx < total; idx += 32) {
    int r = idx / cpr, c = idx - r * cpr;
    *(v8bf*)(dst + r * lstride + c * 8) = *(const v8bf*)(g + (size_t)r * gstride + c * 8);
  }
}
#endif

__global__ void cvt_f32_bf16(const float* __restrict__ src, __bf16* __restrict__ dst, int n) {
  int i = (blockIdx.x * blockDim.x + threadIdx.x) * 4;
  if (i + 3 < n) {
    float4 f = *(const float4*)(src + i);
    dst[i + 0] = f2bf(f.x);
    dst[i + 1] = f2bf(f.y);
    dst[i + 2] = f2bf(f.z);
    dst[i + 3] = f2bf(f.w);
  }
}

// Y = X @ W^T. Each wave: one 16-row m-tile x 64 cols. K loop unrolled x2 with
// alternating register sets: loads for one set issue while the other's WMMAs
// run; no rotation copies.
__global__ __launch_bounds__(128) void proj_wmma(const __bf16* __restrict__ X,  // [M,1024]
                                                 const __bf16* __restrict__ W,  // [N,1024]
                                                 __bf16* __restrict__ Y, int mode) {
  const int lane = threadIdx.x & 31;
  const int wave = threadIdx.x >> 5;
  const int m0 = (blockIdx.x * 4 + wave) * 16;
  const int n0 = blockIdx.y * 64;
  const int row = lane & 15;
  const int ahk = (lane >> 4) ? 8 : 0;
  const int bhk = (lane >> 4) ? 8 : 0;

  const __bf16* xa = X + (size_t)(m0 + row) * DIM_IN + ahk;
  const __bf16* wb = W + (size_t)(n0 + row) * DIM_IN + bhk;

  v8f acc[4] = {(v8f){}, (v8f){}, (v8f){}, (v8f){}};

  // set0 <- k=0
  v16bf a0 = load_a32(xa);
  v16bf b0[4];
#pragma unroll
  for (int f = 0; f < 4; ++f) b0[f] = load_b32(wb + f * 16 * DIM_IN);

  int k = 0;
  for (; k < DIM_IN - 64; k += 64) {
    // prefetch set1 <- k+32
    v16bf a1 = load_a32(xa + k + 32);
    v16bf b1[4];
#pragma unroll
    for (int f = 0; f < 4; ++f) b1[f] = load_b32(wb + f * 16 * DIM_IN + k + 32);
    // consume set0 (k)
#pragma unroll
    for (int f = 0; f < 4; ++f)
      acc[f] = __builtin_amdgcn_wmma_f32_16x16x32_bf16(false, a0, false, b0[f],
                                                       (short)0, acc[f], false, false);
    // prefetch set0 <- k+64
    a0 = load_a32(xa + k + 64);
#pragma unroll
    for (int f = 0; f < 4; ++f) b0[f] = load_b32(wb + f * 16 * DIM_IN + k + 64);
    // consume set1 (k+32)
#pragma unroll
    for (int f = 0; f < 4; ++f)
      acc[f] = __builtin_amdgcn_wmma_f32_16x16x32_bf16(false, a1, false, b1[f],
                                                       (short)0, acc[f], false, false);
  }
  {  // tail: k == DIM_IN-64
    v16bf a1 = load_a32(xa + k + 32);
    v16bf b1[4];
#pragma unroll
    for (int f = 0; f < 4; ++f) b1[f] = load_b32(wb + f * 16 * DIM_IN + k + 32);
#pragma unroll
    for (int f = 0; f < 4; ++f)
      acc[f] = __builtin_amdgcn_wmma_f32_16x16x32_bf16(false, a0, false, b0[f],
                                                       (short)0, acc[f], false, false);
#pragma unroll
    for (int f = 0; f < 4; ++f)
      acc[f] = __builtin_amdgcn_wmma_f32_16x16x32_bf16(false, a1, false, b1[f],
                                                       (short)0, acc[f], false, false);
  }

  const int crow = (lane >> 4) ? 8 : 0;
  const int ncol = lane & 15;
#pragma unroll
  for (int f = 0; f < 4; ++f) {
#pragma unroll
    for (int r = 0; r < 8; ++r) {
      int m = m0 + crow + r;
      int n = n0 + f * 16 + ncol;
      __bf16 val = f2bf(acc[f][r]);
      if (mode == MODE_Q) {           // [B,NH,S,64] head-major
        int bb = m / SEQ, s = m % SEQ, h = n >> 6, d = n & 63;
        Y[(((size_t)bb * NHEAD + h) * SEQ + s) * HDIM + d] = val;
      } else if (mode == MODE_KN) {   // [B*S,64]
        Y[(size_t)m * HDIM + n] = val;
      } else {                        // V transposed: [B,64,S]
        int bb = m / SEQ, s = m % SEQ;
        Y[((size_t)bb * HDIM + n) * SEQ + s] = val;
      }
    }
  }
}

// Flash attention: 4 waves/block, 32 q-rows per wave (128 per block).
// K/V 32-key tiles staged to LDS by TDM (double buffered).
__global__ __launch_bounds__(128) void mqa_flash(const __bf16* __restrict__ Q,   // [B,NH,S,64]
                                                 const __bf16* __restrict__ Kb,  // [B,S,64]
                                                 const __bf16* __restrict__ VT,  // [B,64,S]
                                                 float* __restrict__ Out) {      // [B,S,1024]
  __shared__ __bf16 kbuf[2][32 * KROW];       // 2 x 4608 B
  __shared__ __bf16 vbuf[2][64 * VROW];       // 2 x 5120 B
  __shared__ __bf16 pbuf[4][2][16 * PROW];    // per wave, per row-tile

  const int lane = threadIdx.x & 31;
  const int wave = threadIdx.x >> 5;
  const int bh  = blockIdx.x >> 4;            // 64 (b,h) pairs
  const int grp = blockIdx.x & 15;            // 16 blocks per (b,h)
  const int b = bh >> 4, h = bh & 15;
  const int q0 = (grp * 4 + wave) * 32;

  const __bf16* qbase = Q + (((size_t)b * NHEAD + h) * SEQ) * HDIM;
  const __bf16* kbase = Kb + (size_t)b * SEQ * HDIM;
  const __bf16* vtb   = VT + (size_t)b * HDIM * SEQ;

  const int arow = lane & 15;
  const int ahk = (lane >> 4) ? 8 : 0;
  const int bhk = (lane >> 4) ? 8 : 0;
  const int rbase = (lane >> 4) ? 8 : 0;

  // Q fragments: 2 row-tiles x K chunks {0,32}, resident for the whole kernel.
  v16bf qa[2][2];
#pragma unroll
  for (int t = 0; t < 2; ++t) {
    const __bf16* qrow = qbase + (size_t)(q0 + 16 * t + arow) * HDIM;
    qa[t][0] = load_a32(qrow + 0 + ahk);
    qa[t][1] = load_a32(qrow + 32 + ahk);
  }

  v8f acc[2][4];
  float mrow[2][8], lrow[2][8];
#pragma unroll
  for (int t = 0; t < 2; ++t) {
#pragma unroll
    for (int r = 0; r < 8; ++r) { mrow[t][r] = -__builtin_inff(); lrow[t][r] = 0.0f; }
#pragma unroll
    for (int f = 0; f < 4; ++f) acc[t][f] = (v8f){};
  }

  const float sm = 0.125f * 1.44269504088896f;   // 1/sqrt(64) * log2(e)
  __bf16* pl0 = &pbuf[wave][0][0];
  __bf16* pl1 = &pbuf[wave][1][0];

#if HAS_TDM
  // prologue: DMA tile 0 into buffer 0 (one wave issues; EXEC-independent DMA)
  if (wave == 0) {
    tdm_load_2d((unsigned)(size_t)&kbuf[0][0], kbase, HDIM, SEQ, HDIM, HDIM, 32,
                /*32 DW interval*/4, /*4 DW pad*/3, 1);
    tdm_load_2d((unsigned)(size_t)&vbuf[0][0], vtb, SEQ, HDIM, SEQ, 32, HDIM,
                /*16 DW interval*/3, /*4 DW pad*/3, 1);
  }
#endif

  for (int j = 0; j < SEQ; j += 32) {
    const int cur = (j >> 5) & 1;
#if HAS_TDM
    if (wave == 0) tdm_wait0();       // tile j landed in LDS
    __syncthreads();                  // everyone sees it; prev buffer now free
    if (wave == 0 && (j + 32) < SEQ) {
      const int nxt = cur ^ 1;
      tdm_load_2d((unsigned)(size_t)&kbuf[nxt][0], kbase + (size_t)(j + 32) * HDIM,
                  HDIM, SEQ, HDIM, HDIM, 32, 4, 3, 1);
      tdm_load_2d((unsigned)(size_t)&vbuf[nxt][0], vtb + (j + 32),
                  SEQ, HDIM, SEQ, 32, HDIM, 3, 3, 1);
    }
#else
    __syncthreads();                  // previous readers done
    if (wave == 0) {
      stage_tile(&kbuf[cur][0], kbase + (size_t)j * HDIM, 32, HDIM, HDIM, KROW, lane);
      stage_tile(&vbuf[cur][0], vtb + j, HDIM, 32, SEQ, VROW, lane);
    }
    __syncthreads();
#endif

    // ---- K B-fragments from LDS (shared by both row-tiles) ----
    const __bf16* kl = &kbuf[cur][0];
    v16bf bk00 = load_b32(kl + (0 + arow) * KROW + 0 + bhk);    // keys 0-15, feat 0-31
    v16bf bk01 = load_b32(kl + (0 + arow) * KROW + 32 + bhk);   // keys 0-15, feat 32-63
    v16bf bk10 = load_b32(kl + (16 + arow) * KROW + 0 + bhk);   // keys 16-31
    v16bf bk11 = load_b32(kl + (16 + arow) * KROW + 32 + bhk);

#pragma unroll
    for (int t = 0; t < 2; ++t) {
      v8f c0 = {}, c1 = {};
      c0 = __builtin_amdgcn_wmma_f32_16x16x32_bf16(false, qa[t][0], false, bk00,
                                                   (short)0, c0, false, false);
      c0 = __builtin_amdgcn_wmma_f32_16x16x32_bf16(false, qa[t][1], false, bk01,
                                                   (short)0, c0, false, false);
      c1 = __builtin_amdgcn_wmma_f32_16x16x32_bf16(false, qa[t][0], false, bk10,
                                                   (short)0, c1, false, false);
      c1 = __builtin_amdgcn_wmma_f32_16x16x32_bf16(false, qa[t][1], false, bk11,
                                                   (short)0, c1, false, false);

      __bf16* pl = t ? pl1 : pl0;
#pragma unroll
      for (int r = 0; r < 8; ++r) {
        float s0 = c0[r] * sm, s1 = c1[r] * sm;
        float tm = fmaxf(s0, s1);
#pragma unroll
        for (int off = 1; off < 16; off <<= 1) tm = fmaxf(tm, __shfl_xor(tm, off, 32));
        float mnew  = fmaxf(mrow[t][r], tm);
        float alpha = exp2f(mrow[t][r] - mnew);   // first iter: exp2(-inf)=0
        float p0 = exp2f(s0 - mnew);
        float p1 = exp2f(s1 - mnew);
        lrow[t][r] = lrow[t][r] * alpha + (p0 + p1);  // per-lane partial; reduce at end
        mrow[t][r] = mnew;
        acc[t][0][r] *= alpha; acc[t][1][r] *= alpha;
        acc[t][2][r] *= alpha; acc[t][3][r] *= alpha;
        int prow = rbase + r;
        pl[prow * PROW + (lane & 15)]      = f2bf(p0);
        pl[prow * PROW + 16 + (lane & 15)] = f2bf(p1);
      }
    }

    // ---- V B-fragments from LDS, P A-fragments from LDS, PV WMMAs ----
    const __bf16* vl = &vbuf[cur][0];
    v16bf bv0 = load_b32(vl + (0 * 16 + arow) * VROW + bhk);
    v16bf bv1 = load_b32(vl + (1 * 16 + arow) * VROW + bhk);
    v16bf bv2 = load_b32(vl + (2 * 16 + arow) * VROW + bhk);
    v16bf bv3 = load_b32(vl + (3 * 16 + arow) * VROW + bhk);

#pragma unroll
    for (int t = 0; t < 2; ++t) {
      const __bf16* pr = (t ? pl1 : pl0) + (size_t)arow * PROW + ahk;
      v8bf plo = *(const v8bf*)(pr);
      v8bf phi = *(const v8bf*)(pr + 16);
      v16bf pa;
#pragma unroll
      for (int i = 0; i < 8; ++i) { pa[i] = plo[i]; pa[8 + i] = phi[i]; }
      acc[t][0] = __builtin_amdgcn_wmma_f32_16x16x32_bf16(false, pa, false, bv0,
                                                          (short)0, acc[t][0], false, false);
      acc[t][1] = __builtin_amdgcn_wmma_f32_16x16x32_bf16(false, pa, false, bv1,
                                                          (short)0, acc[t][1], false, false);
      acc[t][2] = __builtin_amdgcn_wmma_f32_16x16x32_bf16(false, pa, false, bv2,
                                                          (short)0, acc[t][2], false, false);
      acc[t][3] = __builtin_amdgcn_wmma_f32_16x16x32_bf16(false, pa, false, bv3,
                                                          (short)0, acc[t][3], false, false);
    }
  }

  // ---- epilogue: reduce per-lane rowsum partials, normalize, store f32 ----
  const int ncol = lane & 15;
#pragma unroll
  for (int t = 0; t < 2; ++t) {
#pragma unroll
    for (int r = 0; r < 8; ++r) {
      float l = lrow[t][r];
#pragma unroll
      for (int off = 1; off < 16; off <<= 1) l += __shfl_xor(l, off, 32);
      float inv = 1.0f / l;
      int s = q0 + 16 * t + rbase + r;
      size_t o = ((size_t)b * SEQ + s) * (NHEAD * HDIM) + (size_t)h * HDIM;
      Out[o + 0 * 16 + ncol] = acc[t][0][r] * inv;
      Out[o + 1 * 16 + ncol] = acc[t][1][r] * inv;
      Out[o + 2 * 16 + ncol] = acc[t][2][r] * inv;
      Out[o + 3 * 16 + ncol] = acc[t][3][r] * inv;
    }
  }
}

extern "C" void kernel_launch(void* const* d_in, const int* in_sizes, int n_in,
                              void* d_out, int out_size, void* d_ws, size_t ws_size,
                              hipStream_t stream) {
  const float* x  = (const float*)d_in[0];
  const float* Wq = (const float*)d_in[1];
  const float* Wk = (const float*)d_in[2];
  const float* Wv = (const float*)d_in[3];

  char* ws = (char*)d_ws;
  __bf16* xb  = (__bf16*)(ws + 0);           // 16,777,216 B
  __bf16* wqb = (__bf16*)(ws + 16777216);    //  2,097,152 B
  __bf16* wkb = (__bf16*)(ws + 18874368);    //    131,072 B
  __bf16* wvb = (__bf16*)(ws + 19005440);    //    131,072 B
  __bf16* qb  = (__bf16*)(ws + 19136512);    // 16,777,216 B
  __bf16* kb  = (__bf16*)(ws + 35913728);    //  1,048,576 B
  __bf16* vtb = (__bf16*)(ws + 36962304);    //  1,048,576 B  (total ~38 MB)

  const int nx = MROWS * DIM_IN;
  cvt_f32_bf16<<<nx / 1024, 256, 0, stream>>>(x, xb, nx);
  cvt_f32_bf16<<<(DIM_IN * DIM_IN) / 1024, 256, 0, stream>>>(Wq, wqb, DIM_IN * DIM_IN);
  cvt_f32_bf16<<<(HDIM * DIM_IN) / 1024, 256, 0, stream>>>(Wk, wkb, HDIM * DIM_IN);
  cvt_f32_bf16<<<(HDIM * DIM_IN) / 1024, 256, 0, stream>>>(Wv, wvb, HDIM * DIM_IN);

  dim3 blk(128);
  proj_wmma<<<dim3(MROWS / 64, DIM_IN / 64), blk, 0, stream>>>(xb, wqb, qb, MODE_Q);
  proj_wmma<<<dim3(MROWS / 64, 1), blk, 0, stream>>>(xb, wkb, kb, MODE_KN);
  proj_wmma<<<dim3(MROWS / 64, 1), blk, 0, stream>>>(xb, wvb, vtb, MODE_VT);

  // 64 (b,h) pairs * 16 blocks, each block = 4 waves * 32 q-rows
  mqa_flash<<<1024, blk, 0, stream>>>(qb, kb, vtb, (float*)d_out);
}
